// HashTrainer_85925115724045
// MI455X (gfx1250) — matrix operand order repaired and checked by reading, not compiled, not run on hardware
//
#include <hip/hip_runtime.h>
#include <hip/hip_bf16.h>
#include <math.h>

#define HKV 8
#define GQAN 4
#define HG 32
#define NI 8192
#define HD 128
#define RB 128
#define NSPLIT 8
#define ROWS_PER_SPLIT (NI / NSPLIT)      // 1024
#define CHUNK 64
#define NCHUNK (ROWS_PER_SPLIT / CHUNK)   // 16

#define WT_S 136   // bf16 stride for W^T in LDS (272B: 16B-aligned rows, bank-spread)
#define KB_S 136   // bf16 stride for K chunk in LDS

#define RSQRT_D 0.08838834764831845f      // 1/sqrt(128)

// workspace layout (floats)
#define WS_MAX      0      // 32
#define WS_SIMPART  32     // 256
#define WS_DECPART  288    // 64
#define WS_QHASH    352    // 4096
#define WS_KSUMPART 4448   // 256*128 = 32768

typedef __attribute__((ext_vector_type(16))) __bf16 v16bf;
typedef __attribute__((ext_vector_type(8)))  __bf16 v8bf;
typedef __attribute__((ext_vector_type(4)))  __bf16 v4bf;
typedef __attribute__((ext_vector_type(8)))  float  v8f;

// tanh(0.1*code) = 2*sigmoid(0.2*code)-1 = 1 - 2/(exp(0.2*code)+1)
__device__ __forceinline__ float hash_act(float code) {
  const float e = __expf(0.2f * code);
  return 1.f - 2.f * __builtin_amdgcn_rcpf(e + 1.f);
}

// ---------------------------------------------------------------------------
// Kernel 1: per-(h,g) max of sim = K.q / sqrt(d)
// ---------------------------------------------------------------------------
__global__ void __launch_bounds__(256)
kmax_kernel(const float* __restrict__ q, const float* __restrict__ k,
            float* __restrict__ ws) {
  __shared__ float qv[HD];
  __shared__ float red[8];
  const int hg = blockIdx.x;
  const int t = threadIdx.x;
  if (t < HD) qv[t] = q[hg * HD + t];
  __syncthreads();
  const float4* kb = (const float4*)(k + (size_t)hg * NI * HD);
  float mx = -1e30f;
  for (int i = 0; i < NI / 256; ++i) {
    const int row = t + 256 * i;
    const float4* kr = kb + (size_t)row * (HD / 4);
    float s = 0.f;
#pragma unroll
    for (int d4 = 0; d4 < HD / 4; ++d4) {
      float4 v = kr[d4];
      s += v.x * qv[4 * d4] + v.y * qv[4 * d4 + 1] +
           v.z * qv[4 * d4 + 2] + v.w * qv[4 * d4 + 3];
    }
    mx = fmaxf(mx, s);
  }
#pragma unroll
  for (int m = 16; m >= 1; m >>= 1) mx = fmaxf(mx, __shfl_xor(mx, m, 32));
  if ((t & 31) == 0) red[t >> 5] = mx;
  __syncthreads();
  if (t == 0) {
    float r = red[0];
    for (int w = 1; w < 8; ++w) r = fmaxf(r, red[w]);
    ws[WS_MAX + hg] = r * RSQRT_D;
  }
}

// ---------------------------------------------------------------------------
// Kernel 2: fused hash-projection GEMM (bf16 WMMA) + hash_sim + sim*exp + colsum
// grid (HG, NSPLIT), 256 threads (8 wave32 waves)
// ---------------------------------------------------------------------------
__global__ void __launch_bounds__(256)
main_kernel(const float* __restrict__ q, const float* __restrict__ k,
            const float* __restrict__ hw, float* __restrict__ ws) {
  __shared__ __bf16 Wt[RB * WT_S];       // W^T[h] as bf16: Wt[r][d]
  __shared__ __bf16 Kb[CHUNK * KB_S];    // K chunk as bf16: Kb[row][d]
  __shared__ float qv[HD];
  __shared__ float qh[RB];
  __shared__ float hash_part[CHUNK * 2]; // [row][colhalf]
  __shared__ float ksum2[RB * 4];        // [col][strip]
  __shared__ float sp[CHUNK];

  const int hg = blockIdx.x;
  const int split = blockIdx.y;
  const int bid = split * HG + hg;
  const int h = hg >> 2;
  const int t = threadIdx.x;
  const int lane = t & 31;
  const int wave = t >> 5;
  const int strip = wave & 3;            // 16-row strip within 64-row chunk
  const int colhalf = wave >> 2;         // which 64 columns of rbit
  const int l16 = lane & 15;
  const int half = lane >> 4;

  if (t < HD) qv[t] = q[hg * HD + t];
  __syncthreads();

  // q_code / q_hash in f32 straight from global W (coalesced per d-row)
  if (t < RB) {
    const float* Wg = hw + h * HD * RB;
    float qc = 0.f;
    for (int d = 0; d < HD; ++d) qc += qv[d] * Wg[d * RB + t];
    const float v = hash_act(qc);
    qh[t] = v;
    if (split == 0) ws[WS_QHASH + hg * RB + t] = v;
  }

  // stage W -> transposed bf16 LDS: Wt[r][d] = W[d][r]
  {
    const float4* Wg4 = (const float4*)(hw + h * HD * RB);
#pragma unroll
    for (int i = 0; i < 16; ++i) {
      const int idx4 = t + 256 * i;            // 4096 float4
      float4 v = Wg4[idx4];
      const int d = idx4 >> 5;
      const int r = (idx4 & 31) * 4;
      Wt[(r + 0) * WT_S + d] = (__bf16)v.x;
      Wt[(r + 1) * WT_S + d] = (__bf16)v.y;
      Wt[(r + 2) * WT_S + d] = (__bf16)v.z;
      Wt[(r + 3) * WT_S + d] = (__bf16)v.w;
    }
  }
  __syncthreads();

  const float m = ws[WS_MAX + hg];
  float colsum[4] = {0.f, 0.f, 0.f, 0.f};
  float simloss = 0.f;

  const float* kbase = k + ((size_t)hg * NI + (size_t)split * ROWS_PER_SPLIT) * HD;

  for (int ch = 0; ch < NCHUNK; ++ch) {
    // stage 64x128 K chunk -> bf16 LDS (coalesced reads, 8B packed LDS stores)
    const float4* src = (const float4*)(kbase + (size_t)ch * CHUNK * HD);
#pragma unroll
    for (int i = 0; i < 8; ++i) {
      const int idx4 = t + 256 * i;            // 2048 float4
      float4 v = src[idx4];
      const int r = idx4 >> 5;
      const int c = (idx4 & 31) * 4;
      v4bf pk = {(__bf16)v.x, (__bf16)v.y, (__bf16)v.z, (__bf16)v.w};
      *(v4bf*)(Kb + r * KB_S + c) = pk;
    }
    // prefetch next chunk into cache while we compute (global_prefetch_b8)
    if (ch + 1 < NCHUNK)
      __builtin_prefetch(kbase + (size_t)(ch + 1) * CHUNK * HD + t * 32, 0, 3);
    __syncthreads();

    // ---- WMMA GEMM: k_code tile (16 rows x 64 cols per wave) ----
    v8f acc[4] = {};
    const __bf16* arowp = Kb + (strip * 16 + l16) * KB_S;
#pragma unroll
    for (int kb = 0; kb < 4; ++kb) {
      const int kb32 = kb * 32;
      // A 16x32 bf16 ISA layout: two contiguous 8-elem runs per lane
      v8bf alo = *(const v8bf*)(arowp + kb32 + 8 * half);
      v8bf ahi = *(const v8bf*)(arowp + kb32 + 16 + 8 * half);
      v16bf a = __builtin_shufflevector(alo, ahi, 0, 1, 2, 3, 4, 5, 6, 7,
                                        8, 9, 10, 11, 12, 13, 14, 15);
#pragma unroll
      for (int c = 0; c < 4; ++c) {
        const int col = colhalf * 64 + c * 16 + l16;
        // B 32x16 bf16 ISA layout: two contiguous 8-elem runs per lane in Wt[col][.]
        const __bf16* bp = Wt + col * WT_S + kb32 + 8 * half;
        v8bf blo = *(const v8bf*)bp;
        v8bf bhi = *(const v8bf*)(bp + 16);
        v16bf b = __builtin_shufflevector(blo, bhi, 0, 1, 2, 3, 4, 5, 6, 7,
                                          8, 9, 10, 11, 12, 13, 14, 15);
        acc[c] = __builtin_amdgcn_wmma_f32_16x16x32_bf16(
            false, a, false, b, (short)0, acc[c], false, false);
      }
    }

    // ---- k_hash, per-row hash distance (deterministic shuffle reduce) ----
    float rs[8];
#pragma unroll
    for (int j = 0; j < 8; ++j) rs[j] = 0.f;
#pragma unroll
    for (int c = 0; c < 4; ++c) {
      const int col = colhalf * 64 + c * 16 + l16;
      const float qhc = qh[col];
#pragma unroll
      for (int j = 0; j < 8; ++j) {
        const float kv = hash_act(acc[c][j]);
        colsum[c] += kv;
        float d = qhc - kv;
        d = d * d;
        d += __shfl_xor(d, 1, 32);
        d += __shfl_xor(d, 2, 32);
        d += __shfl_xor(d, 4, 32);
        d += __shfl_xor(d, 8, 32);   // within-16 group sum over columns
        rs[j] += d;
      }
    }
    if (l16 == 0) {                  // lanes 0 (rows j) and 16 (rows j+8)
      const int rbase = strip * 16 + half * 8;
#pragma unroll
      for (int j = 0; j < 8; ++j)
        hash_part[(rbase + j) * 2 + colhalf] = rs[j];
    }
    __syncthreads();

    // ---- sim = K.q/sqrt(d), accumulate exp(sim-m)*hash_sim ----
    if (t < CHUNK) {
      const __bf16* kr = Kb + t * KB_S;
      float s = 0.f;
#pragma unroll
      for (int g = 0; g < 16; ++g) {
        v8bf v = *(const v8bf*)(kr + g * 8);
#pragma unroll
        for (int e = 0; e < 8; ++e) s += (float)v[e] * qv[g * 8 + e];
      }
      const float e = __expf(s * RSQRT_D - m);
      simloss += e * (hash_part[t * 2] + hash_part[t * 2 + 1]);
    }
    __syncthreads();
  }

  // ---- per-block column sums (deterministic, no atomics) ----
#pragma unroll
  for (int c = 0; c < 4; ++c) {
    float v = colsum[c];
    v += __shfl_xor(v, 16, 32);      // combine the two row-half lanes per column
    if (half == 0)
      ksum2[(colhalf * 64 + c * 16 + lane) * 4 + strip] = v;
  }
  if (t < CHUNK) sp[t] = simloss;
  __syncthreads();
  if (t < RB) {
    const float tot = ksum2[t * 4] + ksum2[t * 4 + 1] +
                      ksum2[t * 4 + 2] + ksum2[t * 4 + 3];
    ws[WS_KSUMPART + bid * RB + t] = tot;
  }
  if (t == 0) {
    float s = 0.f;
    for (int i = 0; i < CHUNK; ++i) s += sp[i];
    ws[WS_SIMPART + bid] = s;
  }
}

// ---------------------------------------------------------------------------
// Kernel 3: decorrelation partials: sum((W^T W - I)^2) per 16-row gram strip
// ---------------------------------------------------------------------------
__global__ void __launch_bounds__(256)
decor_kernel(const float* __restrict__ hw, float* __restrict__ ws) {
  __shared__ float Wc[HD * 16];      // strip's 16 columns of W[h]
  __shared__ float red[256];
  const int b = blockIdx.x;
  const int hh = b >> 3;
  const int strip = b & 7;
  const int t = threadIdx.x;
  const float* Wg = hw + hh * HD * RB;
#pragma unroll
  for (int i = 0; i < 8; ++i) {
    const int e = t + 256 * i;       // d*16 + j
    const int d = e >> 4;
    const int j = e & 15;
    Wc[e] = Wg[d * RB + strip * 16 + j];
  }
  __syncthreads();
  float local = 0.f;
  for (int i = 0; i < 8; ++i) {
    const int e = t + 256 * i;       // 2048 gram entries
    const int rr = e >> 7;
    const int s = e & 127;
    const int r = strip * 16 + rr;
    float dot = 0.f;
    for (int d = 0; d < HD; ++d) dot += Wc[d * 16 + rr] * Wg[d * RB + s];
    const float diff = dot - ((r == s) ? 1.f : 0.f);
    local += diff * diff;
  }
  red[t] = local;
  __syncthreads();
  if (t == 0) {
    float s = 0.f;
    for (int i = 0; i < 256; ++i) s += red[i];
    ws[WS_DECPART + b] = s;
  }
}

// ---------------------------------------------------------------------------
// Kernel 4: combine partials (fixed order) and write the 4 outputs
// ---------------------------------------------------------------------------
__global__ void __launch_bounds__(256)
final_kernel(const float* __restrict__ ws, float* __restrict__ out) {
  __shared__ float red[256];
  const int t = threadIdx.x;
  float local = 0.f;
  for (int i = 0; i < 16; ++i) {
    const int idx = t + 256 * i;     // HG*RB = 4096
    const int hg = idx >> 7;
    const int r = idx & 127;
    float ks = 0.f;
    for (int s = 0; s < NSPLIT; ++s)
      ks += ws[WS_KSUMPART + (s * HG + hg) * RB + r];
    const float bal = (8192.f * ws[WS_QHASH + idx] + ks) * (1.f / (8192.f * 128.f));
    local += bal * bal;
  }
  red[t] = local;
  __syncthreads();
  if (t == 0) {
    float bal_sum = 0.f;
    for (int i = 0; i < 256; ++i) bal_sum += red[i];
    float sim = 0.f;
    for (int i = 0; i < HG * NSPLIT; ++i) sim += ws[WS_SIMPART + i];
    float dec = 0.f;
    for (int i = 0; i < 64; ++i) dec += ws[WS_DECPART + i];
    const float similarity_loss = 1.0f * sim / sqrtf(8192.f);   // EPSILON_
    const float balance_loss = 1.0f * bal_sum;                  // LAMBDA_
    const float decorrelation_loss = 1.0f * sqrtf(dec);         // ETA_
    out[0] = similarity_loss + balance_loss + decorrelation_loss;
    out[1] = similarity_loss;
    out[2] = balance_loss;
    out[3] = decorrelation_loss;
  }
}

// ---------------------------------------------------------------------------
extern "C" void kernel_launch(void* const* d_in, const int* in_sizes, int n_in,
                              void* d_out, int out_size, void* d_ws, size_t ws_size,
                              hipStream_t stream) {
  const float* q  = (const float*)d_in[0];
  const float* k  = (const float*)d_in[1];
  const float* hw = (const float*)d_in[2];
  float* out = (float*)d_out;
  float* ws = (float*)d_ws;

  kmax_kernel<<<HG, 256, 0, stream>>>(q, k, ws);
  main_kernel<<<dim3(HG, NSPLIT), 256, 0, stream>>>(q, k, hw, ws);
  decor_kernel<<<64, 256, 0, stream>>>(hw, ws);
  final_kernel<<<1, 256, 0, stream>>>(ws, out);
}
